// TernaryConv2d_77017353552375
// MI455X (gfx1250) — matrix operand order repaired and checked by reading, not compiled, not run on hardware
//
#include <hip/hip_runtime.h>

typedef __attribute__((ext_vector_type(16))) _Float16 v16h;
typedef __attribute__((ext_vector_type(8)))  _Float16 v8h;
typedef __attribute__((ext_vector_type(8)))  float    v8f;

constexpr int NB    = 32;
constexpr int CIN   = 256;
constexpr int COUT  = 256;
constexpr int HW    = 56;
constexpr int HWHW  = HW * HW;         // 3136
constexpr int NPQ   = NB * HWHW;       // 100352
constexpr int KTOT  = CIN * 9;         // 2304
constexpr int WELEMS = COUT * CIN * 9; // 589824

constexpr int BM = 128;
constexpr int BN = 128;
constexpr int BK = 32;
constexpr int LDSP = 40;               // padded LDS row stride (halfs) -> bank-conflict free
constexpr int KCHUNKS = KTOT / BK;     // 72 (even)

constexpr float THRESH_FACTOR = 0.05f;

// ---------------- workspace layout ----------------
// ws_f[0]            : sum(|w|)      (float)
// ws_u[1]            : nonzero count (unsigned)
// byte offset 256    : ternary weights f16, fragment-ready layout
//                      wta[((k>>5)*COUT + m)*BK + (k&31)],  k = (3r+s)*CIN + c

__global__ void k_init(float* ws) {
    if (threadIdx.x == 0) {
        ws[0] = 0.0f;
        ((unsigned*)ws)[1] = 0u;
    }
}

__global__ __launch_bounds__(256) void k_sumabs(const float* __restrict__ w, float* __restrict__ ws) {
    __shared__ float red[256];
    float s = 0.0f;
    for (int i = blockIdx.x * blockDim.x + threadIdx.x; i < WELEMS; i += gridDim.x * blockDim.x)
        s += fabsf(w[i]);
    red[threadIdx.x] = s;
    __syncthreads();
    for (int off = 128; off > 0; off >>= 1) {
        if ((int)threadIdx.x < off) red[threadIdx.x] += red[threadIdx.x + off];
        __syncthreads();
    }
    if (threadIdx.x == 0) atomicAdd(&ws[0], red[0]);
}

__global__ __launch_bounds__(256) void k_ternarize(const float* __restrict__ w,
                                                   float* __restrict__ ws,
                                                   _Float16* __restrict__ wta) {
    __shared__ unsigned redc[256];
    const float thr = THRESH_FACTOR * ws[0] / (float)WELEMS;
    unsigned cnt = 0;
    for (int i = blockIdx.x * blockDim.x + threadIdx.x; i < WELEMS; i += gridDim.x * blockDim.x) {
        const float v  = w[i];
        const float av = fabsf(v);
        // decode i = ((m*CIN + c)*3 + r)*3 + s
        const int s3 = i % 3;
        const int i3 = i / 3;
        const int r3 = i3 % 3;
        const int i9 = i3 / 3;
        const int c  = i9 % CIN;
        const int m  = i9 / CIN;
        float tv = 0.0f;
        if (av >= thr) tv = (v > 0.0f) ? 1.0f : -1.0f;
        cnt += (tv != 0.0f) ? 1u : 0u;
        const int k  = (r3 * 3 + s3) * CIN + c;       // tap-major, channel within tap
        const int kc = k >> 5;
        const int kk = k & 31;
        wta[(size_t)((kc * COUT + m) * BK + kk)] = (_Float16)tv;  // fragment-ready tiles
    }
    redc[threadIdx.x] = cnt;
    __syncthreads();
    for (int off = 128; off > 0; off >>= 1) {
        if ((int)threadIdx.x < off) redc[threadIdx.x] += redc[threadIdx.x + off];
        __syncthreads();
    }
    if (threadIdx.x == 0) atomicAdd(((unsigned*)ws) + 1, redc[0]);
}

// ------------- implicit-GEMM ternary conv, WMMA f16, double-buffered pipeline -------------
__global__ __launch_bounds__(256) void k_ternconv_wmma(const float* __restrict__ x,
                                                       const _Float16* __restrict__ wta,
                                                       const float* __restrict__ stats,
                                                       float* __restrict__ out) {
    __shared__ _Float16 sB[2 * BN * LDSP];   // double-buffered input-patch tiles

    const int tid  = threadIdx.x;
    const int m0   = blockIdx.y * BM;    // output-channel tile base
    const int pos0 = blockIdx.x * BN;    // flattened (n,p,q) tile base

    const unsigned* su = (const unsigned*)stats;
    const float alpha = stats[0] / ((float)su[1] + 1e-8f);

    // ---- B-load roles (fixed per thread) ----
    const int lrow128 = tid & 127;       // local position row
    const int half_   = tid >> 7;        // which 16 of the 32 k's this thread loads
    const int gpos = pos0 + lrow128;
    const int pn   = gpos / HWHW;
    const int prem = gpos - pn * HWHW;
    const int pp   = prem / HW;
    const int pq   = prem - pp * HW;
    const int xbase_n = pn * CIN * HWHW;

    // ---- wave/lane geometry ----
    const int lane  = tid & 31;
    const int wave  = tid >> 5;
    const int waveM = wave >> 1;         // 0..3 -> 32 M-rows each
    const int waveN = wave & 1;          // 0..1 -> 64 N-cols each
    const int lrow  = lane & 15;
    const int lhi   = lane >> 4;
    const int aRow0 = m0 + waveM * 32 + lrow;        // mi=0 fragment row
    const int aRow1 = aRow0 + 16;                    // mi=1 fragment row

    v8f acc[2][4];
#pragma unroll
    for (int mi = 0; mi < 2; ++mi)
#pragma unroll
        for (int ni = 0; ni < 4; ++ni)
#pragma unroll
            for (int e = 0; e < 8; ++e) acc[mi][ni][e] = 0.0f;

    // ---- pipeline helpers ----
    float bregs[16];

    auto ldB = [&](int kc) {                       // global x -> registers
        const int k0 = kc * BK;
        const int rs = k0 >> 8;
        const int c0 = k0 & 255;
        const int r  = rs / 3;
        const int s  = rs - r * 3;
        const int ih = pp + r - 1;
        const int iw = pq + s - 1;
        const bool valid = ((unsigned)ih < (unsigned)HW) && ((unsigned)iw < (unsigned)HW);
        const int xoff = xbase_n + (c0 + half_ * 16) * HWHW + ih * HW + iw;
#pragma unroll
        for (int i = 0; i < 16; ++i)
            bregs[i] = valid ? x[xoff + i * HWHW] : 0.0f;
    };

    auto stB = [&](int buf) {                      // registers -> LDS (f16, vectorized)
        _Float16* dst = sB + buf * (BN * LDSP) + lrow128 * LDSP + half_ * 16;
        v8h h0, h1;
#pragma unroll
        for (int i = 0; i < 8; ++i) { h0[i] = (_Float16)bregs[i]; h1[i] = (_Float16)bregs[i + 8]; }
        *(v8h*)(dst)     = h0;
        *(v8h*)(dst + 8) = h1;
    };

    auto ldA = [&](int kc, v16h* af) {             // global (L2-hot) -> A fragments, b128
        const _Float16* pa0 = wta + (((size_t)kc * COUT + aRow0) << 5);
        const _Float16* pa1 = wta + (((size_t)kc * COUT + aRow1) << 5);
        v8h a00 = *(const v8h*)(pa0 + 8 * lhi);
        v8h a01 = *(const v8h*)(pa0 + 16 + 8 * lhi);
        v8h a10 = *(const v8h*)(pa1 + 8 * lhi);
        v8h a11 = *(const v8h*)(pa1 + 16 + 8 * lhi);
#pragma unroll
        for (int e = 0; e < 8; ++e) {
            af[0][e] = a00[e]; af[0][e + 8] = a01[e];
            af[1][e] = a10[e]; af[1][e + 8] = a11[e];
        }
    };

    auto compute = [&](int buf, const v16h* af) {  // 8 WMMAs on one K-chunk
        const _Float16* base = sB + buf * (BN * LDSP);
#pragma unroll
        for (int ni = 0; ni < 4; ++ni) {
            const _Float16* pb = base + (waveN * 64 + ni * 16 + lrow) * LDSP + 8 * lhi;
            v8h b0 = *(const v8h*)(pb);
            v8h b1 = *(const v8h*)(pb + 16);
            v16h bf;
#pragma unroll
            for (int e = 0; e < 8; ++e) { bf[e] = b0[e]; bf[e + 8] = b1[e]; }
            acc[0][ni] = __builtin_amdgcn_wmma_f32_16x16x32_f16(
                false, af[0], false, bf, (short)0, acc[0][ni], false, false);
            acc[1][ni] = __builtin_amdgcn_wmma_f32_16x16x32_f16(
                false, af[1], false, bf, (short)0, acc[1][ni], false, false);
        }
    };

    // ---- prologue: chunk 0 staged into buffer 0 ----
    v16h afragE[2], afragO[2];   // fragments for even / odd chunks
    ldB(0);
    stB(0);
    ldA(0, afragE);
    __syncthreads();

    // ---- main pipelined loop, manually unrolled x2 (KCHUNKS is even) ----
    for (int kc = 0; kc < KCHUNKS; kc += 2) {
        const bool more = (kc + 2 < KCHUNKS);

        // even chunk kc: prefetch kc+1, compute buf0, stage into buf1
        ldB(kc + 1);
        ldA(kc + 1, afragO);
        compute(0, afragE);
        stB(1);
        __syncthreads();

        // odd chunk kc+1: prefetch kc+2, compute buf1, stage into buf0
        if (more) ldB(kc + 2);
        if (more) ldA(kc + 2, afragE);
        compute(1, afragO);
        if (more) stB(0);
        __syncthreads();
    }

    // ---- epilogue: C layout = VGPR e -> M = e + 8*lhi, N = lrow ----
#pragma unroll
    for (int ni = 0; ni < 4; ++ni) {
        const int posn = pos0 + waveN * 64 + ni * 16 + lrow;
        const int n  = posn / HWHW;
        const int rm = posn - n * HWHW;
        const int p  = rm / HW;
        const int q  = rm - p * HW;
        const int obase = n * COUT * HWHW + p * HW + q;
#pragma unroll
        for (int mi = 0; mi < 2; ++mi) {
            const int kout0 = m0 + waveM * 32 + mi * 16 + 8 * lhi;
#pragma unroll
            for (int e = 0; e < 8; ++e) {
                out[obase + (kout0 + e) * HWHW] = acc[mi][ni][e] * alpha;
            }
        }
    }
}

extern "C" void kernel_launch(void* const* d_in, const int* in_sizes, int n_in,
                              void* d_out, int out_size, void* d_ws, size_t ws_size,
                              hipStream_t stream) {
    (void)in_sizes; (void)n_in; (void)out_size; (void)ws_size;
    const float* x = (const float*)d_in[0];
    const float* w = (const float*)d_in[1];
    float* out  = (float*)d_out;
    float* ws_f = (float*)d_ws;
    _Float16* wta = (_Float16*)((char*)d_ws + 256);

    k_init<<<1, 64, 0, stream>>>(ws_f);
    k_sumabs<<<288, 256, 0, stream>>>(w, ws_f);
    k_ternarize<<<288, 256, 0, stream>>>(w, ws_f, wta);

    dim3 grid(NPQ / BN, COUT / BM);      // 784 x 2 blocks
    k_ternconv_wmma<<<grid, 256, 0, stream>>>(x, wta, ws_f, out);
}
